// MultiHeadAttentionBlock_22978075033629
// MI455X (gfx1250) — matrix operand order repaired and checked by reading, not compiled, not run on hardware
//
#include <hip/hip_runtime.h>

#define B_ 2
#define S_ 4096
#define D_ 1024
#define H_ 16
#define DK_ 64

typedef __bf16 bf16;
typedef __attribute__((ext_vector_type(16))) __bf16 v16bf;
typedef __attribute__((ext_vector_type(8)))  __bf16 v8bf;
typedef __attribute__((ext_vector_type(8)))  float  v8f;
typedef __attribute__((ext_vector_type(4)))  unsigned u32x4;
typedef __attribute__((ext_vector_type(8)))  int      i32x8;
typedef __attribute__((ext_vector_type(4)))  int      i32x4;

static __device__ __forceinline__ v8f wmma_bf16(v16bf a, v16bf b, v8f c) {
    return __builtin_amdgcn_wmma_f32_16x16x32_bf16(false, a, false, b, (short)0, c,
                                                   false, false);
}

#if __has_builtin(__builtin_amdgcn_tensor_load_to_lds) && \
    __has_builtin(__builtin_amdgcn_s_wait_tensorcnt)
#define HAVE_TDM 1
#endif

// 32-bit LDS byte offset of a __shared__ object (addrspace(3) pointer value).
static __device__ __forceinline__ unsigned lds_offset_of(const void* p) {
    return (unsigned)(unsigned long long)(__attribute__((address_space(3))) const void*)p;
}

#ifdef HAVE_TDM
// Issue a 2D TDM load: tile_d0 x tile_d1 elements (2 bytes each), global row
// stride = stride_elems, packed into LDS at lds_off. D# layout per CDNA5 ISA §8.
static __device__ __forceinline__ void tdm_load_2d_bf16(unsigned lds_off,
                                                        const void* gaddr,
                                                        unsigned tile_d0, unsigned tile_d1,
                                                        unsigned stride_elems) {
    unsigned long long ga = (unsigned long long)gaddr;
    u32x4 g0;
    g0[0] = 1u;                                   // count=1 (valid user descriptor)
    g0[1] = lds_off;                              // lds_addr
    g0[2] = (unsigned)ga;                         // global_addr[31:0]
    g0[3] = (unsigned)(ga >> 32) | (2u << 30);    // global_addr[56:32] | type=2 (image)
    i32x8 g1;
    g1[0] = (int)(1u << 16);                      // workgroup_mask=0, data_size=1 (2B)
    g1[1] = (int)((tile_d0 & 0xFFFFu) << 16);     // tensor_dim0[15:0]
    g1[2] = (int)(((tile_d0 >> 16) & 0xFFFFu) |   // tensor_dim0[31:16]
                  ((tile_d1 & 0xFFFFu) << 16));   // tensor_dim1[15:0]
    g1[3] = (int)(((tile_d1 >> 16) & 0xFFFFu) |   // tensor_dim1[31:16]
                  ((tile_d0 & 0xFFFFu) << 16));   // tile_dim0
    g1[4] = (int)(tile_d1 & 0xFFFFu);             // tile_dim1 (tile_dim2 = 0)
    g1[5] = (int)stride_elems;                    // tensor_dim0_stride[31:0]
    g1[6] = 0;                                    // stride hi / dim1_stride lo
    g1[7] = 0;
    i32x4 z4 = {0, 0, 0, 0};
    i32x8 z8 = {0, 0, 0, 0, 0, 0, 0, 0};
    __builtin_amdgcn_tensor_load_to_lds(g0, g1, z4, z4, z8, 0);
}
#endif

// ---------------------------------------------------------------------------
// One-time fp32 -> bf16 conversion (used for the weight matrices).
// ---------------------------------------------------------------------------
__global__ __launch_bounds__(256) void cvt_f32_bf16_kernel(const float* __restrict__ src,
                                                           bf16* __restrict__ dst) {
    size_t i = ((size_t)blockIdx.x * 256 + threadIdx.x) * 8;
    const float* p = src + i;
    v8bf o;
#pragma unroll
    for (int j = 0; j < 8; ++j) o[j] = (bf16)p[j];
    *(v8bf*)(dst + i) = o;
}

// ---------------------------------------------------------------------------
// GEMM: Y[M,N] = X[M,K] * Wb[N,K]^T   (Wb already bf16)
// Block = 4 waves = 16 x 256 output tile. The 256x32 weight tile is staged
// into LDS by the Tensor Data Mover, double-buffered against compute.
// ---------------------------------------------------------------------------
template <bool IN_BF16, bool OUT_BF16>
__global__ __launch_bounds__(128) void proj_kernel(const void* __restrict__ Xv,
                                                   const bf16* __restrict__ Wb,
                                                   void* __restrict__ Yv,
                                                   int M, int N, int K) {
    __shared__ alignas(64) bf16 Wlds[2][256][32];   // 2 x 16 KB

    const int lane = threadIdx.x & 31;
    const int wave = threadIdx.x >> 5;
    const int hl = lane >> 4;   // half-wave index
    const int ln = lane & 15;
    const int m0 = blockIdx.x * 16;
    const int nblk = blockIdx.y * 256;
    const int arow = m0 + ln;
    const int NI = K / 32;

    unsigned wldsb0 = lds_offset_of(&Wlds[0][0][0]);
    unsigned wldsb1 = lds_offset_of(&Wlds[1][0][0]);

    v8f acc[4] = {};

#ifdef HAVE_TDM
    if (wave == 0)
        tdm_load_2d_bf16(wldsb0, Wb + (size_t)nblk * K, /*d0=*/32, /*d1=*/256, K);
#endif

    for (int i = 0; i < NI; ++i) {
        const int k0 = i * 32;
        const int buf = i & 1;
#ifdef HAVE_TDM
        if (wave == 0) __builtin_amdgcn_s_wait_tensorcnt(0);
        __syncthreads();                         // tile[buf] ready; prev reads done
        if (wave == 0 && i + 1 < NI)
            tdm_load_2d_bf16(buf ? wldsb0 : wldsb1,
                             Wb + (size_t)nblk * K + k0 + 32, 32, 256, K);
#else
        __syncthreads();
        {   // cooperative copy: 256x32 bf16 = 1024 16-byte chunks / 128 threads
#pragma unroll
            for (int t = 0; t < 8; ++t) {
                int e = threadIdx.x + t * 128;
                int r = e >> 2, c = (e & 3) * 8;
                *(v8bf*)&Wlds[buf][r][c] =
                    *(const v8bf*)(Wb + (size_t)(nblk + r) * K + k0 + c);
            }
        }
        __syncthreads();
#endif

        // A fragment (16x32 bf16): lane row = arow, K chunks [k0+hl*16) and [k0+hl*16+8)
        v16bf a;
        if (IN_BF16) {
            const bf16* X = (const bf16*)Xv;
            v8bf c0 = *(const v8bf*)(X + (size_t)arow * K + k0 + hl * 16);
            v8bf c1 = *(const v8bf*)(X + (size_t)arow * K + k0 + 8 + hl * 16);
#pragma unroll
            for (int j = 0; j < 8; ++j) { a[j] = c0[j]; a[8 + j] = c1[j]; }
        } else {
            const float* X = (const float*)Xv;
            const float* p0 = X + (size_t)arow * K + k0 + hl * 16;
            const float* p1 = p0 + 8;
            __builtin_prefetch(p0 + 32, 0, 0);   // global_prefetch_b8
#pragma unroll
            for (int j = 0; j < 8; ++j) { a[j] = (bf16)p0[j]; a[8 + j] = (bf16)p1[j]; }
        }

        // B fragments from LDS: lane = output col, 16 contiguous k values
#pragma unroll
        for (int nt = 0; nt < 4; ++nt) {
            v16bf b = *(const v16bf*)&Wlds[buf][wave * 64 + nt * 16 + ln][hl * 16];
            acc[nt] = wmma_bf16(a, b, acc[nt]);
        }
    }

#pragma unroll
    for (int nt = 0; nt < 4; ++nt) {
        int col = nblk + wave * 64 + nt * 16 + ln;
#pragma unroll
        for (int v = 0; v < 8; ++v) {
            int row = m0 + hl * 8 + v;
            if (OUT_BF16)
                ((bf16*)Yv)[(size_t)row * N + col] = (bf16)acc[nt][v];
            else
                ((float*)Yv)[(size_t)row * N + col] = acc[nt][v];
        }
    }
}

// ---------------------------------------------------------------------------
// Pack mask[1,1,S,S] (int32) into transposed bitmask:
// mpackT[key * (S/32) + qw] bit b  <->  mask[(qw*32+b)*S + key] != 0
// ---------------------------------------------------------------------------
__global__ __launch_bounds__(256) void mask_pack_kernel(const int* __restrict__ mask,
                                                        unsigned* __restrict__ mpackT) {
    int idx = blockIdx.x * 256 + threadIdx.x;     // over S * S/32
    int qw  = idx >> 12;                          // idx / S_
    int key = idx & (S_ - 1);
    const int* mp = mask + (size_t)(qw * 32) * S_ + key;
    unsigned w = 0;
#pragma unroll
    for (int i = 0; i < 32; ++i)
        w |= (mp[(size_t)i * S_] != 0) ? (1u << i) : 0u;
    mpackT[(size_t)key * (S_ / 32) + qw] = w;
}

// ---------------------------------------------------------------------------
// Flash attention. Block = 4 waves; each wave owns 16 query rows of a
// 64-query tile for one (b,h). Streams keys in chunks of 64.
// K tile staged to LDS by the Tensor Data Mover; V tile transposed into LDS.
// ---------------------------------------------------------------------------
__global__ __launch_bounds__(128) void attn_kernel(const bf16* __restrict__ Qb,
                                                   const bf16* __restrict__ Kb,
                                                   const bf16* __restrict__ Vb,
                                                   const unsigned* __restrict__ mpackT,
                                                   bf16* __restrict__ Cb) {
    __shared__ alignas(64) bf16 Klds[64][DK_];     // K tile: [key][dk]     (8 KB)
    __shared__ alignas(64) bf16 VT[DK_][64];       // V^T tile: [dk][key]   (8 KB)
    __shared__ alignas(64) bf16 Plds[4][16][64];   // per-wave P tile       (8 KB)

    const int lane = threadIdx.x & 31;
    const int wave = threadIdx.x >> 5;
    const int hl = lane >> 4;
    const int ln = lane & 15;

    const int bh = blockIdx.y;            // 0 .. B*H-1
    const int b  = bh / H_;
    const int h  = bh % H_;
    const int qrow0 = blockIdx.x * 64 + wave * 16;

    const bf16* Qbh = Qb + (size_t)b * S_ * D_ + h * DK_;
    const bf16* Kbh = Kb + (size_t)b * S_ * D_ + h * DK_;
    const bf16* Vbh = Vb + (size_t)b * S_ * D_ + h * DK_;

    // Q A-fragments (16 x 64 = two 16x32 frags), loaded once
    v16bf aq0, aq1;
    {
        const bf16* qp = Qbh + (size_t)(qrow0 + ln) * D_;
        v8bf c;
        c = *(const v8bf*)(qp + hl * 16);
#pragma unroll
        for (int i = 0; i < 8; ++i) aq0[i] = c[i];
        c = *(const v8bf*)(qp + 8 + hl * 16);
#pragma unroll
        for (int i = 0; i < 8; ++i) aq0[8 + i] = c[i];
        c = *(const v8bf*)(qp + 32 + hl * 16);
#pragma unroll
        for (int i = 0; i < 8; ++i) aq1[i] = c[i];
        c = *(const v8bf*)(qp + 40 + hl * 16);
#pragma unroll
        for (int i = 0; i < 8; ++i) aq1[8 + i] = c[i];
    }

    float mrow[8], lrow[8];
    v8f co[4] = {};
#pragma unroll
    for (int v = 0; v < 8; ++v) { mrow[v] = -3.0e38f; lrow[v] = 0.f; }

    const int bitbase = (qrow0 + hl * 8) & 31;
    const int qword   = (qrow0 + hl * 8) >> 5;

    for (int key0 = 0; key0 < S_; key0 += 64) {
        __syncthreads();   // previous iteration's LDS reads complete

        // ---- stage K tile via Tensor Data Mover (one DMA per block) ----
#ifdef HAVE_TDM
        if (wave == 0) {
            tdm_load_2d_bf16(lds_offset_of(&Klds[0][0]),
                             Kbh + (size_t)key0 * D_,
                             /*tile_d0=*/DK_, /*tile_d1=*/64,
                             /*stride=*/D_);
            __builtin_amdgcn_s_wait_tensorcnt(0);
        }
#else
        {
            const bf16* gsrc = Kbh + (size_t)key0 * D_;
#pragma unroll
            for (int i = 0; i < 4; ++i) {
                int e = threadIdx.x + i * 128;   // 16-byte chunk index (512 total)
                int r = e >> 3, c8 = e & 7;
                *(v8bf*)&Klds[r][c8 * 8] = *(const v8bf*)(gsrc + (size_t)r * D_ + c8 * 8);
            }
        }
#endif

        // ---- stage V^T tile cooperatively (128 threads) ----
        {
            int r  = threadIdx.x >> 1;            // key row 0..63
            int ch = (threadIdx.x & 1) * 32;      // dk half
            const bf16* vp = Vbh + (size_t)(key0 + r) * D_ + ch;
#pragma unroll
            for (int i = 0; i < 32; ++i) VT[ch + i][r] = vp[i];
        }
        __syncthreads();

        // ---- scores: S = Q * K^T for 4 tiles of 16 keys (K frags from LDS) ----
        v8f sc[4];
#pragma unroll
        for (int nt = 0; nt < 4; ++nt) {
            const bf16* kp = &Klds[nt * 16 + ln][hl * 16];
            v16bf bk0 = *(const v16bf*)(kp);
            v16bf bk1 = *(const v16bf*)(kp + 32);
            v8f s = {};
            s = wmma_bf16(aq0, bk0, s);
            s = wmma_bf16(aq1, bk1, s);
            sc[nt] = s;
        }

        // ---- scale + mask, row max ----
        float cmax[8];
#pragma unroll
        for (int v = 0; v < 8; ++v) cmax[v] = -3.0e38f;
#pragma unroll
        for (int nt = 0; nt < 4; ++nt) {
            int key = key0 + nt * 16 + ln;
            unsigned mw = mpackT[(size_t)key * (S_ / 32) + qword];
#pragma unroll
            for (int v = 0; v < 8; ++v) {
                float x = sc[nt][v] * 0.125f;                 // 1/sqrt(64)
                x = ((mw >> (bitbase + v)) & 1u) ? x : -1.0e9f;
                sc[nt][v] = x;
                cmax[v] = fmaxf(cmax[v], x);
            }
        }
#pragma unroll
        for (int v = 0; v < 8; ++v) {
            float t = cmax[v];
#pragma unroll
            for (int off = 1; off < 16; off <<= 1)
                t = fmaxf(t, __shfl_xor(t, off, 32));
            cmax[v] = t;
        }

        // ---- online softmax update ----
        float alpha[8], csum[8];
#pragma unroll
        for (int v = 0; v < 8; ++v) {
            float mn = fmaxf(mrow[v], cmax[v]);
            alpha[v] = __expf(mrow[v] - mn);
            mrow[v] = mn;
            csum[v] = 0.f;
        }
#pragma unroll
        for (int nt = 0; nt < 4; ++nt)
#pragma unroll
            for (int v = 0; v < 8; ++v) {
                float p = __expf(sc[nt][v] - mrow[v]);
                sc[nt][v] = p;
                csum[v] += p;
            }
#pragma unroll
        for (int v = 0; v < 8; ++v) {
            float t = csum[v];
#pragma unroll
            for (int off = 1; off < 16; off <<= 1)
                t += __shfl_xor(t, off, 32);
            lrow[v] = lrow[v] * alpha[v] + t;
        }
#pragma unroll
        for (int nt = 0; nt < 4; ++nt)
#pragma unroll
            for (int v = 0; v < 8; ++v) co[nt][v] *= alpha[v];

        // ---- P (C-layout) -> LDS -> A-layout ----
#pragma unroll
        for (int nt = 0; nt < 4; ++nt)
#pragma unroll
            for (int v = 0; v < 8; ++v)
                Plds[wave][hl * 8 + v][nt * 16 + ln] = (bf16)sc[nt][v];

        v16bf ap0, ap1;
        {
            const bf16* pp = &Plds[wave][ln][0];
            v8bf c;
            c = *(const v8bf*)(pp + hl * 16);
#pragma unroll
            for (int i = 0; i < 8; ++i) ap0[i] = c[i];
            c = *(const v8bf*)(pp + 8 + hl * 16);
#pragma unroll
            for (int i = 0; i < 8; ++i) ap0[8 + i] = c[i];
            c = *(const v8bf*)(pp + 32 + hl * 16);
#pragma unroll
            for (int i = 0; i < 8; ++i) ap1[i] = c[i];
            c = *(const v8bf*)(pp + 40 + hl * 16);
#pragma unroll
            for (int i = 0; i < 8; ++i) ap1[8 + i] = c[i];
        }

        // ---- ctx += P(16x64) * V(64x64) ----
#pragma unroll
        for (int nt = 0; nt < 4; ++nt) {
            const bf16* vtp = &VT[nt * 16 + ln][0];
            v16bf bv0 = *(const v16bf*)(vtp + hl * 16);
            v16bf bv1 = *(const v16bf*)(vtp + 32 + hl * 16);
            co[nt] = wmma_bf16(ap0, bv0, co[nt]);
            co[nt] = wmma_bf16(ap1, bv1, co[nt]);
        }
    }

    // ---- normalize and store ctx (bf16, [B,S,D] with head-interleaved cols) ----
#pragma unroll
    for (int nt = 0; nt < 4; ++nt) {
        int col = h * DK_ + nt * 16 + ln;
#pragma unroll
        for (int v = 0; v < 8; ++v) {
            int row = qrow0 + hl * 8 + v;
            float o = co[nt][v] / lrow[v];
            Cb[((size_t)b * S_ + row) * D_ + col] = (bf16)o;
        }
    }
}

// ---------------------------------------------------------------------------
extern "C" void kernel_launch(void* const* d_in, const int* in_sizes, int n_in,
                              void* d_out, int out_size, void* d_ws, size_t ws_size,
                              hipStream_t stream) {
    const float* q    = (const float*)d_in[0];
    const float* k    = (const float*)d_in[1];
    const float* v    = (const float*)d_in[2];
    const int*   mask = (const int*)d_in[3];
    const float* w_q  = (const float*)d_in[4];
    const float* w_k  = (const float*)d_in[5];
    const float* w_v  = (const float*)d_in[6];
    const float* w_o  = (const float*)d_in[7];
    float* out = (float*)d_out;

    const size_t n_elem = (size_t)B_ * S_ * D_;   // 8M
    const size_t w_elem = (size_t)D_ * D_;        // 1M
    bf16* Qb = (bf16*)d_ws;
    bf16* Kb = Qb + n_elem;
    bf16* Vb = Kb + n_elem;
    bf16* Cb = Vb + n_elem;
    unsigned* mpackT = (unsigned*)(Cb + n_elem);           // 2 MB
    bf16* Wqb = (bf16*)(mpackT + (size_t)S_ * (S_ / 32));
    bf16* Wkb = Wqb + w_elem;
    bf16* Wvb = Wkb + w_elem;
    bf16* Wob = Wvb + w_elem;

    const int M = B_ * S_;
    dim3 blk(128);
    dim3 gproj(M / 16, D_ / 256);                  // 512 x 4
    int gcvt = (int)(w_elem / (256 * 8));          // 512 blocks

    cvt_f32_bf16_kernel<<<gcvt, 256, 0, stream>>>(w_q, Wqb);
    cvt_f32_bf16_kernel<<<gcvt, 256, 0, stream>>>(w_k, Wkb);
    cvt_f32_bf16_kernel<<<gcvt, 256, 0, stream>>>(w_v, Wvb);
    cvt_f32_bf16_kernel<<<gcvt, 256, 0, stream>>>(w_o, Wob);

    proj_kernel<false, true><<<gproj, blk, 0, stream>>>(q, Wqb, Qb, M, D_, D_);
    proj_kernel<false, true><<<gproj, blk, 0, stream>>>(k, Wkb, Kb, M, D_, D_);
    proj_kernel<false, true><<<gproj, blk, 0, stream>>>(v, Wvb, Vb, M, D_, D_);

    mask_pack_kernel<<<(S_ * (S_ / 32)) / 256, 256, 0, stream>>>(mask, mpackT);

    dim3 gattn(S_ / 64, B_ * H_);                  // 64 x 32
    attn_kernel<<<gattn, blk, 0, stream>>>(Qb, Kb, Vb, mpackT, Cb);

    proj_kernel<true, false><<<gproj, blk, 0, stream>>>(Cb, Wob, out, M, D_, D_);
}
